// CausalSelfAttention_90168543412719
// MI455X (gfx1250) — matrix operand order repaired and checked by reading, not compiled, not run on hardware
//
#include <hip/hip_runtime.h>

// ---------------------------------------------------------------------------
// Types for CDNA5 WMMA (wave32, 16x16x32 f16 -> f32)
// ---------------------------------------------------------------------------
typedef __attribute__((ext_vector_type(16))) _Float16 v16h;
typedef __attribute__((ext_vector_type(8)))  _Float16 v8h;
typedef __attribute__((ext_vector_type(8)))  float    v8f;

union F16Frag { v16h v; v8h h[2]; _Float16 e[16]; };

__device__ __forceinline__ v8f wmma16(v16h a, v16h b, v8f c) {
  return __builtin_amdgcn_wmma_f32_16x16x32_f16(false, a, false, b, (short)0, c,
                                                false, false);
}

// ---------------------------------------------------------------------------
// Generic tiled GEMM: C[M,N] = act(A[M,K] @ B[K,N] + bias)
// fp32 in/out, f16 WMMA inside. Block 256 threads = 8 waves.
// Block tile 128x128, wave tile 32(M) x 64(N), K-step 32.
// Interior tiles stage via float4 loads; edge tiles use guarded scalar loads.
// act: 0 = none, 1 = SiLU
// ---------------------------------------------------------------------------
__global__ __launch_bounds__(256)
void gemm_tile_f16(const float* __restrict__ A, const float* __restrict__ B,
                   float* __restrict__ C, int M, int N, int K,
                   int lda, int ldb, int ldc,
                   const float* __restrict__ bias, int act) {
  __shared__ __align__(32) _Float16 As[128 * 32];   // [m][k]
  __shared__ __align__(32) _Float16 Bs[128 * 32];   // [n][k] (transposed)
  const int tid  = threadIdx.x;
  const int lane = tid & 31;
  const int wave = tid >> 5;
  const int wm   = wave & 3;     // 4 waves along M
  const int wn   = wave >> 2;    // 2 waves along N
  const int lh   = lane & 15;
  const int hlf  = lane >> 4;
  const int kbA  = hlf * 8;
  const int bm   = blockIdx.y * 128;
  const int bn   = blockIdx.x * 128;

  const v8f vzero = {0.f, 0.f, 0.f, 0.f, 0.f, 0.f, 0.f, 0.f};
  v8f acc[2][4];
#pragma unroll
  for (int i = 0; i < 2; ++i)
#pragma unroll
    for (int j = 0; j < 4; ++j) acc[i][j] = vzero;

  for (int k0 = 0; k0 < K; k0 += 32) {
    // Prefetch next K-slab (gfx1250 global_prefetch_b8 path)
    if (k0 + 32 < K) {
      int pr = bm + (tid & 127);
      if (pr < M) __builtin_prefetch(&A[(long)pr * lda + k0 + 32]);
      int pc = bn + (tid & 127);
      if (pc < N) __builtin_prefetch(&B[(long)(k0 + 32 + (tid >> 7) * 16) * ldb + pc]);
    }
    // ---- Stage A tile (128x32) as f16 ----
    if (bm + 128 <= M && k0 + 32 <= K) {
#pragma unroll
      for (int i = 0; i < 4; ++i) {
        int idx = tid + i * 256;          // 1024 quads = 128 rows x 8 quads
        int r = idx >> 3;
        int c4 = (idx & 7) * 4;
        const float4 f = *(const float4*)&A[(long)(bm + r) * lda + k0 + c4];
        As[r * 32 + c4 + 0] = (_Float16)f.x;
        As[r * 32 + c4 + 1] = (_Float16)f.y;
        As[r * 32 + c4 + 2] = (_Float16)f.z;
        As[r * 32 + c4 + 3] = (_Float16)f.w;
      }
    } else {
#pragma unroll
      for (int i = 0; i < 16; ++i) {
        int idx = tid + i * 256;
        int r = idx >> 5, c = idx & 31;
        int gr = bm + r, gc = k0 + c;
        float vv = 0.f;
        if (gr < M && gc < K) vv = A[(long)gr * lda + gc];
        As[r * 32 + c] = (_Float16)vv;
      }
    }
    // ---- Stage B tile (32x128) transposed into [n][k] ----
    if (bn + 128 <= N && k0 + 32 <= K) {
#pragma unroll
      for (int i = 0; i < 4; ++i) {
        int idx = tid + i * 256;          // 1024 quads = 32 k-rows x 32 quads
        int kk = idx >> 5;
        int c4 = (idx & 31) * 4;
        const float4 f = *(const float4*)&B[(long)(k0 + kk) * ldb + bn + c4];
        Bs[(c4 + 0) * 32 + kk] = (_Float16)f.x;
        Bs[(c4 + 1) * 32 + kk] = (_Float16)f.y;
        Bs[(c4 + 2) * 32 + kk] = (_Float16)f.z;
        Bs[(c4 + 3) * 32 + kk] = (_Float16)f.w;
      }
    } else {
#pragma unroll
      for (int i = 0; i < 16; ++i) {
        int idx = tid + i * 256;
        int kk = idx >> 7, n = idx & 127;
        int gk = k0 + kk, gn = bn + n;
        float vv = 0.f;
        if (gk < K && gn < N) vv = B[(long)gk * ldb + gn];
        Bs[n * 32 + kk] = (_Float16)vv;
      }
    }
    __syncthreads();

    // A fragments: lane row = lh, elements K = hlf*16 + 0..15 (contiguous)
    v16h af[2];
#pragma unroll
    for (int mi = 0; mi < 2; ++mi) {
      int row = wm * 32 + mi * 16 + lh;
      F16Frag u;
      u.v = *(const v16h*)&As[row * 32 + hlf * 16];
      af[mi] = u.v;
    }
    // B fragments: lane col = lh, elements K = hlf*16 + 0..15 (contiguous)
    v16h bf[4];
#pragma unroll
    for (int ni = 0; ni < 4; ++ni) {
      int col = wn * 64 + ni * 16 + lh;
      bf[ni] = *(const v16h*)&Bs[col * 32 + hlf * 16];
    }
#pragma unroll
    for (int mi = 0; mi < 2; ++mi)
#pragma unroll
      for (int ni = 0; ni < 4; ++ni)
        acc[mi][ni] = wmma16(af[mi], bf[ni], acc[mi][ni]);
    __syncthreads();
  }

  // Epilogue: C/D layout -> lane = N col, VGPR j = row j (+8 for upper half)
#pragma unroll
  for (int mi = 0; mi < 2; ++mi)
#pragma unroll
    for (int ni = 0; ni < 4; ++ni)
#pragma unroll
      for (int j = 0; j < 8; ++j) {
        int row = bm + wm * 32 + mi * 16 + j + hlf * 8;
        int col = bn + wn * 64 + ni * 16 + lh;
        if (row < M && col < N) {
          float vv = acc[mi][ni][j];
          if (bias) vv += bias[col];
          if (act == 1) vv = vv / (1.f + __expf(-vv));
          C[(long)row * ldc + col] = vv;
        }
      }
}

// ---------------------------------------------------------------------------
// Flash attention on pre-converted f16 operands:
//   Qh [B,H,qlen,128] f16 (row-major head dim)
//   Kh: rows = keys (klenPad rows x 128), contiguous head dim   -> B frags
//   Vt: transposed, [128, klenPad] per head, contiguous keys    -> B frags
// klenPad is a multiple of 32; rows >= klen are zero-filled (masked anyway).
// One wave handles 16 query rows. Online softmax; QK^T and PV on WMMA.
// O[idx] = (accum ? O[idx] : 0) + gate * softmax(QK^T)V
// ---------------------------------------------------------------------------
__global__ __launch_bounds__(32)
void flash_f16(const _Float16* __restrict__ Qh, const _Float16* __restrict__ Kh,
               const _Float16* __restrict__ Vt, float* __restrict__ O,
               int qlen, int klen, int klenPad,
               long kvBatchStride, long kvHeadStride,
               int causal, const float* __restrict__ gatep, int accum) {
  __shared__ __align__(32) _Float16 Pl[16 * 32];
  const int lane = threadIdx.x & 31;
  const int lh   = lane & 15;
  const int hlf  = lane >> 4;
  const int kbA  = hlf * 8;
  const int b  = blockIdx.z;
  const int h  = blockIdx.y;
  const int qt = blockIdx.x;
  const int nheads = gridDim.y;
  const long qbase = ((long)(b * nheads + h) * qlen + qt * 16) * 128;
  const _Float16* kb = Kh + (long)b * kvBatchStride + (long)h * kvHeadStride;
  const _Float16* vb = Vt + (long)b * kvBatchStride + (long)h * kvHeadStride;
  const float scale = 0.08838834764831845f;  // 1/sqrt(128)

  // Q fragments (A layout) for the 4 x 32-wide chunks of HS=128
  v16h aq[4];
  {
    const _Float16* qp = Qh + qbase + (long)lh * 128;
#pragma unroll
    for (int c = 0; c < 4; ++c) {
      F16Frag u;
      u.v = *(const v16h*)&qp[c * 32 + hlf * 16];
      aq[c] = u.v;
    }
  }

  const v8f vzero = {0.f, 0.f, 0.f, 0.f, 0.f, 0.f, 0.f, 0.f};
  v8f oacc[8];
  float m_i[8], l_i[8];
#pragma unroll
  for (int i = 0; i < 8; ++i) { oacc[i] = vzero; m_i[i] = -1e30f; l_i[i] = 0.f; }

  int nkt = (klen + 31) / 32;
  if (causal) { int lim = (qt * 16 + 16 + 31) / 32; if (lim < nkt) nkt = lim; }

  for (int kt = 0; kt < nkt; ++kt) {
    float sv[2][8];
#pragma unroll
    for (int g = 0; g < 2; ++g) {
      int key = kt * 32 + g * 16 + lh;       // always < klenPad (valid memory)
      const _Float16* kr = kb + (long)key * 128;
      v8f s = vzero;
#pragma unroll
      for (int c = 0; c < 4; ++c)            // B frag: contiguous 16 halves
        s = wmma16(aq[c], *(const v16h*)&kr[c * 32 + hlf * 16], s);
#pragma unroll
      for (int j = 0; j < 8; ++j) {
        int row = qt * 16 + j + hlf * 8;
        int col = kt * 32 + g * 16 + lh;
        float x = s[j] * scale;
        if (col >= klen || (causal && col > row)) x = -1e30f;
        sv[g][j] = x;
      }
    }
    // Online softmax update (reduce across 16-lane halves; rows j / j+8)
    float alpha[8];
#pragma unroll
    for (int j = 0; j < 8; ++j) {
      float r = fmaxf(sv[0][j], sv[1][j]);
      r = fmaxf(r, __shfl_xor(r, 1, 16));
      r = fmaxf(r, __shfl_xor(r, 2, 16));
      r = fmaxf(r, __shfl_xor(r, 4, 16));
      r = fmaxf(r, __shfl_xor(r, 8, 16));
      float mn = fmaxf(m_i[j], r);
      alpha[j] = __expf(m_i[j] - mn);
      float ps = 0.f;
#pragma unroll
      for (int g = 0; g < 2; ++g) {
        float p = (sv[g][j] <= -1e29f) ? 0.f : __expf(sv[g][j] - mn);
        sv[g][j] = p;
        ps += p;
      }
      ps += __shfl_xor(ps, 1, 16);
      ps += __shfl_xor(ps, 2, 16);
      ps += __shfl_xor(ps, 4, 16);
      ps += __shfl_xor(ps, 8, 16);
      l_i[j] = l_i[j] * alpha[j] + ps;
      m_i[j] = mn;
    }
#pragma unroll
    for (int nt = 0; nt < 8; ++nt)
#pragma unroll
      for (int j = 0; j < 8; ++j) oacc[nt][j] *= alpha[j];

    // Transpose P (D layout) -> A layout via LDS
    __syncthreads();
#pragma unroll
    for (int g = 0; g < 2; ++g)
#pragma unroll
      for (int j = 0; j < 8; ++j)
        Pl[(j + hlf * 8) * 32 + g * 16 + lh] = (_Float16)sv[g][j];
    __syncthreads();
    v16h pa;
    {
      F16Frag u;
      u.v = *(const v16h*)&Pl[lh * 32 + hlf * 16];
      pa = u.v;
    }
    // O += P(16x32) @ V(32x128): Vt gives contiguous keys per lane column
#pragma unroll
    for (int nt = 0; nt < 8; ++nt) {
      const _Float16* vr = vb + (long)(nt * 16 + lh) * klenPad + kt * 32 + hlf * 16;
      oacc[nt] = wmma16(pa, *(const v16h*)vr, oacc[nt]);
    }
  }

  float gate = gatep ? gatep[0] : 1.0f;
#pragma unroll
  for (int nt = 0; nt < 8; ++nt)
#pragma unroll
    for (int j = 0; j < 8; ++j) {
      int row = qt * 16 + j + hlf * 8;
      long oi = ((long)(b * nheads + h) * qlen + row) * 128 + nt * 16 + lh;
      float vv = (l_i[j] > 0.f) ? oacc[nt][j] / l_i[j] : 0.f;
      float prev = accum ? O[oi] : 0.f;
      O[oi] = prev + gate * vv;
    }
}

// ---------------------------------------------------------------------------
// Elementwise / reduction helper kernels
// ---------------------------------------------------------------------------
__global__ __launch_bounds__(256)
void rmsnorm_kernel(const float* __restrict__ in, const float* __restrict__ w,
                    float* __restrict__ out, int cols) {
  __shared__ float red[256];
  int tid = threadIdx.x;
  long row = blockIdx.x;
  const float* r = in + row * cols;
  float s = 0.f;
  for (int c = tid; c < cols; c += 256) { float t = r[c]; s += t * t; }
  red[tid] = s;
  __syncthreads();
  for (int st = 128; st > 0; st >>= 1) {
    if (tid < st) red[tid] += red[tid + st];
    __syncthreads();
  }
  float inv = rsqrtf(red[0] / (float)cols + 1e-5f);
  for (int c = tid; c < cols; c += 256) out[row * cols + c] = r[c] * inv * w[c];
}

// qkv[B,T,3C] -> q(f32, for q2 GEMM), qh/kh (f16 [B,NH,T,HS]) with RoPE,
// vt (f16 transposed [B,NH,HS,T])
__global__ __launch_bounds__(256)
void rope_split_kernel(const float* __restrict__ qkv, const float* __restrict__ cs,
                       const float* __restrict__ sn, float* __restrict__ q,
                       _Float16* __restrict__ qh, _Float16* __restrict__ kh,
                       _Float16* __restrict__ vt) {
  long i = (long)blockIdx.x * 256 + threadIdx.x;  // over B*T*NH*(HS/2)
  if (i >= 4194304L) return;
  int d2 = (int)(i & 63);
  long r = i >> 6;
  int h = (int)(r & 31); r >>= 5;
  int t = (int)(r & 1023); r >>= 10;
  int b = (int)r;
  long src = ((long)(b * 1024 + t)) * 12288 + h * 128 + d2 * 2;
  float c = cs[t * 64 + d2], s = sn[t * 64 + d2];
  float q0 = qkv[src],        q1 = qkv[src + 1];
  float k0 = qkv[src + 4096], k1 = qkv[src + 4097];
  float v0 = qkv[src + 8192], v1 = qkv[src + 8193];
  float qr0 = q0 * c - q1 * s, qr1 = q1 * c + q0 * s;
  float kr0 = k0 * c - k1 * s, kr1 = k1 * c + k0 * s;
  long dst = (((long)(b * 32 + h)) * 1024 + t) * 128 + d2 * 2;
  q[dst] = qr0;              q[dst + 1] = qr1;
  qh[dst] = (_Float16)qr0;   qh[dst + 1] = (_Float16)qr1;
  kh[dst] = (_Float16)kr0;   kh[dst + 1] = (_Float16)kr1;
  long vtd = (((long)(b * 32 + h)) * 128 + d2 * 2) * 1024 + t;
  vt[vtd] = (_Float16)v0;    vt[vtd + 1024] = (_Float16)v1;
}

// aqkv[10,12288] -> akh f16 [NH,32(pad),128], avt f16 [NH,128,32(pad)]
__global__ __launch_bounds__(256)
void adapter_split_kernel(const float* __restrict__ aqkv,
                          _Float16* __restrict__ akh, _Float16* __restrict__ avt) {
  int i = blockIdx.x * 256 + threadIdx.x;  // NH*32*HS = 131072
  if (i >= 131072) return;
  int d = i & 127;
  int t = (i >> 7) & 31;
  int h = i >> 12;
  float kv = 0.f, vv = 0.f;
  if (t < 10) {
    kv = aqkv[t * 12288 + 4096 + h * 128 + d];
    vv = aqkv[t * 12288 + 8192 + h * 128 + d];
  }
  akh[(h * 32 + t) * 128 + d] = (_Float16)kv;
  avt[(h * 128 + d) * 32 + t] = (_Float16)vv;
}

// pkh f16 [B,NH,1536(pad),128]: keys row-major
__global__ __launch_bounds__(256)
void build_pk_kernel(const float* __restrict__ wflat,
                     const float* __restrict__ pad, _Float16* __restrict__ pkh) {
  long i = (long)blockIdx.x * 256 + threadIdx.x;  // B*NH*1536*128
  if (i >= 12582912L) return;
  int d = (int)(i & 127);
  long r = i >> 7;
  int t = (int)(r % 1536); r /= 1536;
  int h = (int)(r & 31);
  int b = (int)(r >> 5);
  float val = 0.f;
  if (t < 1500) {
    if (h < 20 && d < 64) val = wflat[(long)b * 1920000 + (long)h * 96000 + t * 64 + d];
    else val = pad[((long)b * 1500 + t) * 128 + d];
  }
  pkh[i] = (_Float16)val;
}

// pvt f16 [B,NH,128,1536(pad)]: transposed (contiguous keys)
__global__ __launch_bounds__(256)
void build_pvt_kernel(const float* __restrict__ wflat,
                      const float* __restrict__ pad, _Float16* __restrict__ pvt) {
  long i = (long)blockIdx.x * 256 + threadIdx.x;  // B*NH*1536*128
  if (i >= 12582912L) return;
  int d = (int)(i & 127);
  long r = i >> 7;
  int t = (int)(r % 1536); r /= 1536;
  int h = (int)(r & 31);
  int b = (int)(r >> 5);
  float val = 0.f;
  if (t < 1500) {
    if (h < 20 && d < 64) val = wflat[(long)b * 1920000 + (long)h * 96000 + t * 64 + d];
    else val = pad[((long)b * 1500 + t) * 128 + d];
  }
  pvt[(((long)(b * 32 + h)) * 128 + d) * 1536 + t] = (_Float16)val;
}

// q2h[b,g,t,d] = (f16) sum_h q2a[b,h,t,d] * w[h,g]
__global__ __launch_bounds__(256)
void headmix_kernel(const float* __restrict__ qa, const float* __restrict__ w,
                    _Float16* __restrict__ qo) {
  long i = (long)blockIdx.x * 256 + threadIdx.x;  // B*NH*T*HS
  if (i >= 8388608L) return;
  int d = (int)(i & 127);
  long r = i >> 7;
  int t = (int)(r & 1023); r >>= 10;
  int g = (int)(r & 31);
  int b = (int)(r >> 5);
  float s = 0.f;
  for (int h = 0; h < 32; ++h)
    s += qa[(((long)(b * 32 + h)) * 1024 + t) * 128 + d] * w[h * 32 + g];
  qo[i] = (_Float16)s;
}

// y [B,NH,T,HS] -> [B,T,C]
__global__ __launch_bounds__(256)
void to_btc_kernel(const float* __restrict__ yin, float* __restrict__ yout) {
  long i = (long)blockIdx.x * 256 + threadIdx.x;  // B*T*C
  if (i >= 8388608L) return;
  int d = (int)(i & 127);
  long r = i >> 7;
  int h = (int)(r & 31); r >>= 5;
  int t = (int)(r & 1023);
  int b = (int)(r >> 10);
  yout[i] = yin[(((long)(b * 32 + h)) * 1024 + t) * 128 + d];
}

// ---------------------------------------------------------------------------
// Host-side orchestration
// ---------------------------------------------------------------------------
extern "C" void kernel_launch(void* const* d_in, const int* in_sizes, int n_in,
                              void* d_out, int out_size, void* d_ws, size_t ws_size,
                              hipStream_t stream) {
  (void)in_sizes; (void)n_in; (void)out_size; (void)ws_size;
  const float* x      = (const float*)d_in[0];
  const float* audio  = (const float*)d_in[1];
  const float* rcos   = (const float*)d_in[2];
  const float* rsin   = (const float*)d_in[3];
  // d_in[4] = mask (causal, reimplemented), d_in[5] = max_seq_length (unused)
  const float* pbk    = (const float*)d_in[6];
  const float* pbv    = (const float*)d_in[7];
  const float* c_attn = (const float*)d_in[8];
  const float* c_proj = (const float*)d_in[9];
  const float* awte   = (const float*)d_in[10];
  const float* gatef  = (const float*)d_in[11];
  const float* rms_g  = (const float*)d_in[12];
  const float* rms_k  = (const float*)d_in[13];
  const float* rms_v  = (const float*)d_in[14];
  const float* pdown  = (const float*)d_in[15];
  const float* pup    = (const float*)d_in[16];
  const float* pq128  = (const float*)d_in[17];
  const float* pq32   = (const float*)d_in[18];
  const float* pgate  = (const float*)d_in[19];
  const float* wkw    = (const float*)d_in[20];
  const float* wvw    = (const float*)d_in[21];
  const float* wvb    = (const float*)d_in[22];
  float* out = (float*)d_out;

  // Workspace layout (256B-aligned regions)
  char* base = (char*)d_ws;
  size_t cur = 0;
  auto alloc = [&](size_t bytes) {
    char* p = base + cur;
    cur += (bytes + 255) & ~(size_t)255;
    return p;
  };
  float*    qkv  = (float*)   alloc(100663296);  // [B,T,3C]; reused by pkh/pvt
  float*    q    = (float*)   alloc(33554432);   // f32 q for q2 GEMM
  float*    y    = (float*)   alloc(33554432);   // accumulated attention out
  float*    yc   = (float*)   alloc(33554432);   // [B,T,C]
  float*    q2a  = (float*)   alloc(33554432);
  _Float16* qh   = (_Float16*)alloc(16777216);
  _Float16* kh   = (_Float16*)alloc(16777216);
  _Float16* vt   = (_Float16*)alloc(16777216);
  _Float16* q2h  = (_Float16*)alloc(16777216);
  float*    wkl  = (float*)   alloc(15360000);
  float*    wvl  = (float*)   alloc(15360000);
  float*    wkm  = (float*)   alloc(960000);
  float*    wvm  = (float*)   alloc(960000);
  float*    wk2  = (float*)   alloc(15360000);
  float*    wv2  = (float*)   alloc(15360000);
  float*    pre  = (float*)   alloc(163840);
  float*    aqkv = (float*)   alloc(491520);
  _Float16* akh  = (_Float16*)alloc(262144);
  _Float16* avt  = (_Float16*)alloc(262144);
  // pkh/pvt reuse the qkv region (qkv is consumed by rope_split before these)
  _Float16* pkh  = (_Float16*)qkv;                 // 25,165,824 B
  _Float16* pvt  = (_Float16*)((char*)qkv + 25165824);

  dim3 blk(256);

  // 1) qkv = x @ c_attn_w     (2048 x 12288 x 4096)
  gemm_tile_f16<<<dim3(96, 16), blk, 0, stream>>>(x, c_attn, qkv, 2048, 12288, 4096,
                                                  4096, 12288, 12288, nullptr, 0);
  // 2) RoPE + split: q (f32), qh/kh (f16), vt (f16 transposed)
  rope_split_kernel<<<16384, blk, 0, stream>>>(qkv, rcos, rsin, q, qh, kh, vt);
  // 3) y = causal SDPA(q,k,v)
  flash_f16<<<dim3(64, 32, 2), dim3(32), 0, stream>>>(
      qh, kh, vt, y, 1024, 1024, 1024,
      (long)32 * 1024 * 128, (long)1024 * 128, 1, nullptr, 0);
  // 4) adapter: prefix = rmsnorm(adapter_wte); aqkv = prefix @ c_attn_w
  rmsnorm_kernel<<<10, blk, 0, stream>>>(awte, rms_g, pre, 4096);
  gemm_tile_f16<<<dim3(96, 1), blk, 0, stream>>>(pre, c_attn, aqkv, 10, 12288, 4096,
                                                 4096, 12288, 12288, nullptr, 0);
  adapter_split_kernel<<<512, blk, 0, stream>>>(aqkv, akh, avt);
  // 5) y += gating_factor * SDPA(q, ak, av)   (K/V batch-broadcast, pad 10->32)
  flash_f16<<<dim3(64, 32, 2), dim3(32), 0, stream>>>(
      qh, akh, avt, y, 1024, 10, 32, 0L, (long)32 * 128, 0, gatef, 1);
  // 6) whisper key path: linear -> rmsnorm -> silu(down) -> up
  gemm_tile_f16<<<dim3(10, 24), blk, 0, stream>>>(audio, wkw, wkl, 3000, 1280, 1280,
                                                  1280, 1280, 1280, nullptr, 0);
  rmsnorm_kernel<<<3000, blk, 0, stream>>>(wkl, rms_k, wkl, 1280);
  gemm_tile_f16<<<dim3(1, 24), blk, 0, stream>>>(wkl, pdown, wkm, 3000, 80, 1280,
                                                 1280, 80, 80, nullptr, 1);
  gemm_tile_f16<<<dim3(10, 24), blk, 0, stream>>>(wkm, pup, wk2, 3000, 1280, 80,
                                                  80, 1280, 1280, nullptr, 0);
  // 7) whisper value path (bias on first linear)
  gemm_tile_f16<<<dim3(10, 24), blk, 0, stream>>>(audio, wvw, wvl, 3000, 1280, 1280,
                                                  1280, 1280, 1280, wvb, 0);
  rmsnorm_kernel<<<3000, blk, 0, stream>>>(wvl, rms_v, wvl, 1280);
  gemm_tile_f16<<<dim3(1, 24), blk, 0, stream>>>(wvl, pdown, wvm, 3000, 80, 1280,
                                                 1280, 80, 80, nullptr, 1);
  gemm_tile_f16<<<dim3(10, 24), blk, 0, stream>>>(wvm, pup, wv2, 3000, 1280, 80,
                                                  80, 1280, 1280, nullptr, 0);
  // 8) assemble pk (row-major) / pvt (key-contiguous), pad 1500->1536
  build_pk_kernel<<<49152, blk, 0, stream>>>(wk2, pbk, pkh);
  build_pvt_kernel<<<49152, blk, 0, stream>>>(wv2, pbv, pvt);
  // 9) q2 = headmix(q @ proj_q128, proj_q32) -> f16
  gemm_tile_f16<<<dim3(1, 512), blk, 0, stream>>>(q, pq128, q2a, 65536, 128, 128,
                                                  128, 128, 128, nullptr, 0);
  headmix_kernel<<<32768, blk, 0, stream>>>(q2a, pq32, q2h);
  // 10) y += proj_gating * SDPA(q2, pk, pv)
  flash_f16<<<dim3(64, 32, 2), dim3(32), 0, stream>>>(
      q2h, pkh, pvt, y, 1024, 1500, 1536,
      (long)32 * 1536 * 128, (long)1536 * 128, 0, pgate, 1);
  // 11) transpose to [B,T,C], 12) out = yc @ c_proj_w
  to_btc_kernel<<<32768, blk, 0, stream>>>(y, yc);
  gemm_tile_f16<<<dim3(32, 16), blk, 0, stream>>>(yc, c_proj, out, 2048, 4096, 4096,
                                                  4096, 4096, 4096, nullptr, 0);
}
